// EncoderOnlyTransformer_70153995813314
// MI455X (gfx1250) — compile-verified, matrix operand order checked by reading, not run on hardware
//
#include <hip/hip_runtime.h>
#include <hip/hip_bf16.h>
#include <math.h>

#define LYR  12
#define NH   12
#define CDIM 768
#define VDIM 8192
#define TDIM 1024
#define BDIM 8
#define HD   64
#define MROWS (BDIM*TDIM) /* 8192 */

typedef __bf16 bf16x16 __attribute__((ext_vector_type(16)));
typedef float  f32x8   __attribute__((ext_vector_type(8)));
typedef unsigned int u32x4 __attribute__((ext_vector_type(4)));
typedef int          i32x8 __attribute__((ext_vector_type(8)));
typedef int          i32x4 __attribute__((ext_vector_type(4)));

__device__ __forceinline__ unsigned short f2bf(float f) {
  unsigned int u = __float_as_uint(f);
  u += 0x7FFFu + ((u >> 16) & 1u);   // round-to-nearest-even
  return (unsigned short)(u >> 16);
}

// TDM: load a 2D tile (tileK elems wide, 256 rows) of a bf16 row-major tensor
// with row stride K into LDS at ldsoff (packed tileK elems/row).
__device__ __forceinline__ void tdm_load_b(const unsigned short* gsrc,
                                           unsigned ldsoff, int K, unsigned tileK) {
  unsigned long long ga = (unsigned long long)(uintptr_t)gsrc;
  u32x4 g0;
  g0[0] = 1u;                                        // count=1, user mode
  g0[1] = ldsoff;                                    // lds_addr (bytes)
  g0[2] = (unsigned)(ga & 0xffffffffu);              // global_addr[31:0]
  g0[3] = (unsigned)((ga >> 32) & 0x01ffffffu) | 0x80000000u;  // addr[56:32]|type=2
  unsigned dim0 = (unsigned)K;                       // tensor_dim0
  unsigned dim1 = 256u;                              // tensor_dim1
  unsigned long long s0 = (unsigned long long)K;     // tensor_dim0_stride
  i32x8 g1;
  g1[0] = (int)(1u << 16);                           // data_size=1 (2 bytes)
  g1[1] = (int)((dim0 & 0xffffu) << 16);             // dim0[15:0] @ bits 63:48
  g1[2] = (int)((dim0 >> 16) | ((dim1 & 0xffffu) << 16));
  g1[3] = (int)((dim1 >> 16) | (tileK << 16));       // tile_dim0 = tileK
  g1[4] = (int)(256u);                               // tile_dim1 = 256, tile_dim2=0
  g1[5] = (int)(s0 & 0xffffffffu);
  g1[6] = (int)((s0 >> 32) & 0xffffu);               // stride1 = 0
  g1[7] = 0;
  i32x4 z4 = {0, 0, 0, 0};
  i32x8 z8 = {0, 0, 0, 0, 0, 0, 0, 0};
  __builtin_amdgcn_tensor_load_to_lds(g0, g1, z4, z4, z8, 0);
}

// ---------------- embedding: x = wte[idx] + wpe ----------------
__global__ __launch_bounds__(256) void k_embed(const int* __restrict__ idx,
                                               const float* __restrict__ wte,
                                               const float* __restrict__ wpe,
                                               float* __restrict__ x) {
  int e = blockIdx.x * 256 + threadIdx.x;           // e < MROWS*CDIM
  int row = e / CDIM, c = e - row * CDIM;
  int t = row & (TDIM - 1);
  x[e] = wte[(long long)idx[row] * CDIM + c] + wpe[(long long)t * CDIM + c];
}

// ------------- weight convert: W[K,N] f32 -> Wt[N,K] bf16 -------------
__global__ __launch_bounds__(256) void k_convT(const float* __restrict__ W,
                                               unsigned short* __restrict__ Wt,
                                               int K, int N) {
  long long base = (long long)blockIdx.y * K * N;
  int e = blockIdx.x * 256 + threadIdx.x;
  if (e >= K * N) return;
  int k = e / N, n = e - k * N;
  Wt[base + (long long)n * K + k] = f2bf(W[base + e]);
}

// -------- V transpose: qkvb[row, 2C + h*64+d] -> vt[b,h,d, t] --------
__global__ __launch_bounds__(256) void k_vtrans(const unsigned short* __restrict__ qkvb,
                                                unsigned short* __restrict__ vt) {
  int e = blockIdx.x * 256 + threadIdx.x;   // e < MROWS*CDIM
  int row = e / CDIM, c = e - row * CDIM;   // c = h*64+d
  int b = row / TDIM, t = row - b * TDIM;
  int h = c / HD, d = c - h * HD;
  vt[(long long)((b * NH + h) * HD + d) * TDIM + t] =
      qkvb[(long long)row * (3 * CDIM) + 2 * CDIM + c];
}

// ---------------- layernorm (wave per row) -> bf16 ----------------
__global__ __launch_bounds__(256) void k_ln(const float* __restrict__ x,
                                            const float* __restrict__ w,
                                            const float* __restrict__ b,
                                            unsigned short* __restrict__ out) {
  int wave = threadIdx.x >> 5, lane = threadIdx.x & 31;
  int row = blockIdx.x * 8 + wave;
  const float* xr = x + (long long)row * CDIM;
  float s = 0.f, s2 = 0.f;
  for (int c = lane; c < CDIM; c += 32) { float v = xr[c]; s += v; s2 += v * v; }
  for (int o = 16; o >= 1; o >>= 1) { s += __shfl_xor(s, o, 32); s2 += __shfl_xor(s2, o, 32); }
  float mean = s * (1.f / CDIM);
  float var  = s2 * (1.f / CDIM) - mean * mean;
  float rstd = rsqrtf(var + 1e-5f);
  unsigned short* orow = out + (long long)row * CDIM;
  for (int c = lane; c < CDIM; c += 32)
    orow[c] = f2bf((xr[c] - mean) * rstd * w[c] + b[c]);
}

// ---------------- WMMA flash attention ----------------
__global__ __launch_bounds__(256) void k_fattn(const unsigned short* __restrict__ qkvb,
                                               const unsigned short* __restrict__ vt,
                                               unsigned short* __restrict__ y) {
  __shared__ alignas(16) unsigned short plds[8][16][32];   // per-wave P bounce (8KB)
  union FragAB { bf16x16 v; uint4 q[2]; };
  int wave = threadIdx.x >> 5, lane = threadIdx.x & 31;
  int half = lane >> 4, r = lane & 15;
  int b = blockIdx.z, h = blockIdx.y;
  int q0 = (blockIdx.x * 8 + wave) * 16;
  const int QL = 3 * CDIM;  // 2304

  const unsigned short* Qrow = qkvb + (long long)(b * TDIM + q0 + r) * QL + h * HD;
  FragAB qa0, qa1;
  qa0.q[0] = *(const uint4*)(Qrow + half * 8);
  qa0.q[1] = *(const uint4*)(Qrow + 16 + half * 8);
  qa1.q[0] = *(const uint4*)(Qrow + 32 + half * 8);
  qa1.q[1] = *(const uint4*)(Qrow + 48 + half * 8);

  f32x8 accO[4] = {};
  float mrow[8], lrow[8];
  #pragma unroll
  for (int i = 0; i < 8; ++i) { mrow[i] = -1e30f; lrow[i] = 0.f; }
  const unsigned short* Vth = vt + (long long)((b * NH + h) * HD) * TDIM;

  for (int j = 0; j < TDIM; j += 32) {
    const unsigned short* Krow0 = qkvb + (long long)(b * TDIM + j + r) * QL + CDIM + h * HD;
    const unsigned short* Krow1 = Krow0 + (long long)16 * QL;
    FragAB kb;
    f32x8 s0 = {}, s1 = {};
    kb.q[0] = *(const uint4*)(Krow0 + half * 8);
    kb.q[1] = *(const uint4*)(Krow0 + 16 + half * 8);
    s0 = __builtin_amdgcn_wmma_f32_16x16x32_bf16(false, qa0.v, false, kb.v, (short)0, s0, false, false);
    kb.q[0] = *(const uint4*)(Krow0 + 32 + half * 8);
    kb.q[1] = *(const uint4*)(Krow0 + 48 + half * 8);
    s0 = __builtin_amdgcn_wmma_f32_16x16x32_bf16(false, qa1.v, false, kb.v, (short)0, s0, false, false);
    kb.q[0] = *(const uint4*)(Krow1 + half * 8);
    kb.q[1] = *(const uint4*)(Krow1 + 16 + half * 8);
    s1 = __builtin_amdgcn_wmma_f32_16x16x32_bf16(false, qa0.v, false, kb.v, (short)0, s1, false, false);
    kb.q[0] = *(const uint4*)(Krow1 + 32 + half * 8);
    kb.q[1] = *(const uint4*)(Krow1 + 48 + half * 8);
    s1 = __builtin_amdgcn_wmma_f32_16x16x32_bf16(false, qa1.v, false, kb.v, (short)0, s1, false, false);

    #pragma unroll
    for (int i = 0; i < 8; ++i) {
      float v0 = s0[i] * 0.125f, v1 = s1[i] * 0.125f;
      float tm = fmaxf(v0, v1);
      for (int o = 8; o >= 1; o >>= 1) tm = fmaxf(tm, __shfl_xor(tm, o, 32));
      float nm = fmaxf(mrow[i], tm);
      float corr = __expf(mrow[i] - nm);
      float p0 = __expf(v0 - nm), p1 = __expf(v1 - nm);
      float ts = p0 + p1;
      for (int o = 8; o >= 1; o >>= 1) ts += __shfl_xor(ts, o, 32);
      lrow[i] = lrow[i] * corr + ts;
      mrow[i] = nm;
      #pragma unroll
      for (int t = 0; t < 4; ++t) accO[t][i] *= corr;
      plds[wave][half * 8 + i][r]      = f2bf(p0);
      plds[wave][half * 8 + i][16 + r] = f2bf(p1);
    }
    FragAB pa;
    pa.q[0] = *(const uint4*)(&plds[wave][r][half * 8]);
    pa.q[1] = *(const uint4*)(&plds[wave][r][16 + half * 8]);
    #pragma unroll
    for (int t = 0; t < 4; ++t) {
      const unsigned short* Vrow = Vth + (long long)(t * 16 + r) * TDIM + j;
      FragAB vb;
      vb.q[0] = *(const uint4*)(Vrow + half * 8);
      vb.q[1] = *(const uint4*)(Vrow + 16 + half * 8);
      accO[t] = __builtin_amdgcn_wmma_f32_16x16x32_bf16(false, pa.v, false, vb.v, (short)0, accO[t], false, false);
    }
  }

  #pragma unroll
  for (int t = 0; t < 4; ++t) {
    int d = t * 16 + r;
    #pragma unroll
    for (int i = 0; i < 8; ++i) {
      int row = q0 + half * 8 + i;
      y[(long long)(b * TDIM + row) * CDIM + h * HD + d] = f2bf(accO[t][i] / lrow[i]);
    }
  }
}

// ---------------- WMMA bf16 GEMM with TDM-staged B tiles ----------------
// out[M,N] = A[M,K](bf16) * Wt[N,K]^T(bf16)  + epilogue
// EPI: 1 = +bias+residual -> f32, 2 = gelu(+bias) -> bf16, 3 = plain -> f32, 4 = +bias -> bf16
template <int EPI>
__global__ __launch_bounds__(256) void k_gemm(const unsigned short* __restrict__ A,
                                              const unsigned short* __restrict__ Wt,
                                              const float* __restrict__ bias,
                                              const float* __restrict__ res,
                                              float* __restrict__ outF,
                                              unsigned short* __restrict__ outH,
                                              int M, int N, int K) {
  // double-buffered B tile: 256 rows (n) x 64 k, packed 64 elems/row (64KB)
  __shared__ alignas(16) unsigned short Btile[2][256 * 64];
  union Frag { bf16x16 v; uint4 q[2]; };
  int tid = threadIdx.x;
  int wave = tid >> 5, lane = tid & 31;
  int wm = wave >> 2, wn = wave & 3;
  int half = lane >> 4, r = lane & 15;
  int m0 = blockIdx.y * 32 + wm * 16;
  int n0 = blockIdx.x * 256 + wn * 64;
  int nblk = blockIdx.x * 256;
  const unsigned short* Arow = A + (long long)(m0 + r) * K;
  unsigned lds0 = (unsigned)(uintptr_t)&Btile[0][0];
  unsigned lds1 = (unsigned)(uintptr_t)&Btile[1][0];

  // prologue: stage first B tile via Tensor Data Mover
  if (wave == 0) {
    tdm_load_b(Wt + (long long)nblk * K, lds0, K, 64u);
    __builtin_amdgcn_s_wait_tensorcnt(0);
  }
  __syncthreads();

  f32x8 acc[4] = {};
  int p = 0;
  for (int k = 0; k < K; k += 64) {
    // issue DMA for the next tile while computing on this one
    if (k + 64 < K && wave == 0)
      tdm_load_b(Wt + (long long)nblk * K + (k + 64), p ? lds0 : lds1, K, 64u);

    __builtin_prefetch(Arow + k + 128, 0, 1);   // speculative global_prefetch (A stream)
    #pragma unroll
    for (int kk = 0; kk < 64; kk += 32) {
      Frag a;
      a.q[0] = *(const uint4*)(Arow + k + kk + half * 8);
      a.q[1] = *(const uint4*)(Arow + k + kk + 16 + half * 8);
      #pragma unroll
      for (int j = 0; j < 4; ++j) {
        const unsigned short* Bl = &Btile[p][(wn * 64 + j * 16 + r) * 64 + kk];
        Frag bq;
        bq.q[0] = *(const uint4*)(Bl + half * 8);
        bq.q[1] = *(const uint4*)(Bl + 16 + half * 8);
        acc[j] = __builtin_amdgcn_wmma_f32_16x16x32_bf16(
            false, a.v, false, bq.v, (short)0, acc[j], false, false);
      }
    }

    if (k + 64 < K) {
      if (wave == 0) __builtin_amdgcn_s_wait_tensorcnt(0);
      __syncthreads();
      p ^= 1;
    }
  }

  #pragma unroll
  for (int j = 0; j < 4; ++j) {
    int n = n0 + j * 16 + r;
    float bv = (EPI == 3) ? 0.f : bias[n];
    #pragma unroll
    for (int i = 0; i < 8; ++i) {
      int row = m0 + half * 8 + i;
      long long o = (long long)row * N + n;
      float v = acc[j][i] + bv;
      if (EPI == 1) v += res[o];
      if (EPI == 2) {
        float t = v + 0.044715f * v * v * v;
        v = 0.5f * v * (1.f + tanhf(0.7978845608028654f * t));
        outH[o] = f2bf(v);
      } else if (EPI == 4) {
        outH[o] = f2bf(v);
      } else {
        outF[o] = v;
      }
    }
  }
}

extern "C" void kernel_launch(void* const* d_in, const int* in_sizes, int n_in,
                              void* d_out, int out_size, void* d_ws, size_t ws_size,
                              hipStream_t stream) {
  (void)in_sizes; (void)n_in; (void)out_size; (void)ws_size;
  const int*   idx   = (const int*)d_in[0];
  const float* wte   = (const float*)d_in[1];
  const float* wpe   = (const float*)d_in[2];
  const float* ln1w  = (const float*)d_in[3];
  const float* ln1b  = (const float*)d_in[4];
  const float* Wqkv  = (const float*)d_in[5];
  const float* bqkv  = (const float*)d_in[6];
  const float* Wproj = (const float*)d_in[7];
  const float* bproj = (const float*)d_in[8];
  const float* ln2w  = (const float*)d_in[9];
  const float* ln2b  = (const float*)d_in[10];
  const float* Wfc   = (const float*)d_in[11];
  const float* bfc   = (const float*)d_in[12];
  const float* Wfc2  = (const float*)d_in[13];
  const float* bfc2  = (const float*)d_in[14];
  const float* lnfw  = (const float*)d_in[15];
  const float* lnfb  = (const float*)d_in[16];
  const float* Whead = (const float*)d_in[17];

  char* ws = (char*)d_ws;
  float*          xf     = (float*)(ws + 0);                       // 8192*768 f32
  unsigned short* qkvb   = (unsigned short*)(ws + 25165824LL);     // 8192*2304 bf16
  unsigned short* vt     = (unsigned short*)(ws + 62914560LL);     // 8*12*64*1024 bf16
  unsigned short* hb     = (unsigned short*)(ws + 75497472LL);     // 8192*768 bf16
  unsigned short* yb     = (unsigned short*)(ws + 88080384LL);     // 8192*768 bf16
  unsigned short* gb     = (unsigned short*)(ws + 100663296LL);    // 8192*3072 bf16
  unsigned short* WtQKV  = (unsigned short*)(ws + 150994944LL);    // 12*768*2304
  unsigned short* WtProj = (unsigned short*)(ws + 193462272LL);    // 12*768*768
  unsigned short* WtFc   = (unsigned short*)(ws + 207618048LL);    // 12*768*3072
  unsigned short* WtFc2  = (unsigned short*)(ws + 264241152LL);    // 12*3072*768
  unsigned short* WtHead = (unsigned short*)(ws + 320864256LL);    // 768*8192

  k_convT<<<dim3((CDIM*3*CDIM+255)/256, LYR), 256, 0, stream>>>(Wqkv,  WtQKV,  CDIM,   3*CDIM);
  k_convT<<<dim3((CDIM*CDIM+255)/256,   LYR), 256, 0, stream>>>(Wproj, WtProj, CDIM,   CDIM);
  k_convT<<<dim3((CDIM*4*CDIM+255)/256, LYR), 256, 0, stream>>>(Wfc,   WtFc,   CDIM,   4*CDIM);
  k_convT<<<dim3((4*CDIM*CDIM+255)/256, LYR), 256, 0, stream>>>(Wfc2,  WtFc2,  4*CDIM, CDIM);
  k_convT<<<dim3((CDIM*VDIM+255)/256,   1  ), 256, 0, stream>>>(Whead, WtHead, CDIM,   VDIM);

  k_embed<<<MROWS*CDIM/256, 256, 0, stream>>>(idx, wte, wpe, xf);

  for (int l = 0; l < LYR; ++l) {
    k_ln<<<MROWS/8, 256, 0, stream>>>(xf, ln1w + l*CDIM, ln1b + l*CDIM, hb);
    k_gemm<4><<<dim3(3*CDIM/256, MROWS/32), 256, 0, stream>>>(
        hb, WtQKV + (size_t)l*CDIM*3*CDIM, bqkv + (size_t)l*3*CDIM,
        nullptr, nullptr, qkvb, MROWS, 3*CDIM, CDIM);
    k_vtrans<<<MROWS*CDIM/256, 256, 0, stream>>>(qkvb, vt);
    k_fattn<<<dim3(TDIM/128, NH, BDIM), 256, 0, stream>>>(qkvb, vt, yb);
    k_gemm<1><<<dim3(CDIM/256, MROWS/32), 256, 0, stream>>>(
        yb, WtProj + (size_t)l*CDIM*CDIM, bproj + (size_t)l*CDIM,
        xf, xf, nullptr, MROWS, CDIM, CDIM);
    k_ln<<<MROWS/8, 256, 0, stream>>>(xf, ln2w + l*CDIM, ln2b + l*CDIM, hb);
    k_gemm<2><<<dim3(4*CDIM/256, MROWS/32), 256, 0, stream>>>(
        hb, WtFc + (size_t)l*CDIM*4*CDIM, bfc + (size_t)l*4*CDIM,
        nullptr, nullptr, gb, MROWS, 4*CDIM, CDIM);
    k_gemm<1><<<dim3(CDIM/256, MROWS/32), 256, 0, stream>>>(
        gb, WtFc2 + (size_t)l*4*CDIM*CDIM, bfc2 + (size_t)l*CDIM,
        xf, xf, nullptr, MROWS, CDIM, 4*CDIM);
  }
  k_ln<<<MROWS/8, 256, 0, stream>>>(xf, lnfw, lnfb, hb);
  k_gemm<3><<<dim3(VDIM/256, MROWS/32), 256, 0, stream>>>(
      hb, WtHead, nullptr, nullptr, (float*)d_out, nullptr, MROWS, VDIM, CDIM);
}
